// OuterProductMean_63763084476749
// MI455X (gfx1250) — matrix operand order repaired and checked by reading, not compiled, not run on hardware
//
#include <hip/hip_runtime.h>
#include <hip/hip_bf16.h>

// ---------- constants ----------
#define N_SEQ 128
#define N_RES 256
#define C_M   256
#define C_C   32     // projection channels
#define C_Z   128    // output channels
#define KOUT  1024   // C*C flattened

// LDS padding (strides chosen so row-stride % 256B leaves lanes on distinct banks)
#define BT_ROW_B   272           // 128 bf16 row padded to 136 bf16 = 272 B
#define BT_TILE_B  (512 * BT_ROW_B)          // 16 j * 32 d rows  = 139264 B
#define A2_ROW_B   2064          // 1024 bf16 row padded to 1032 bf16 = 2064 B
#define A2_SLICE_B (16 * A2_ROW_B)           // 16 pairs per wave  = 33024 B
#define LDS_TOTAL  (BT_TILE_B + 4 * A2_SLICE_B)   // 271360 B

typedef __bf16 v16bf __attribute__((ext_vector_type(16)));
typedef float  v8f   __attribute__((ext_vector_type(8)));

union Frag16 {            // 16 bf16 = 32 bytes = one WMMA A/B operand
    v16bf v;
    uint4 u[2];
};
union Pack8 { __bf16 h[8]; uint4 q; };

static __device__ __forceinline__ v8f wmma_bf16(const Frag16& a, const Frag16& b, v8f c) {
    return __builtin_amdgcn_wmma_f32_16x16x32_bf16(false, a.v, false, b.v,
                                                   (short)0, c, false, false);
}

// =====================================================================
// Kernel 1: LayerNorm + dual 256->32 projections, bf16 outputs
//   aT[(i*32+c)*128 + s],  bT[(j*32+d)*128 + s]
// one block per (s,i), 256 threads
// =====================================================================
__global__ void opm_ln_proj_kernel(const float* __restrict__ m,
                                   const float* __restrict__ ln_w,
                                   const float* __restrict__ ln_b,
                                   const float* __restrict__ w1,
                                   const float* __restrict__ b1,
                                   const float* __restrict__ w2,
                                   const float* __restrict__ b2,
                                   __bf16* __restrict__ aT,
                                   __bf16* __restrict__ bT) {
    __shared__ float red[C_M];
    __shared__ float mnS[C_M];

    const int s = blockIdx.x;
    const int i = blockIdx.y;
    const int t = threadIdx.x;

    const float x = m[((size_t)s * N_RES + i) * C_M + t];

    red[t] = x;
    __syncthreads();
    for (int st = C_M / 2; st > 0; st >>= 1) {
        if (t < st) red[t] += red[t + st];
        __syncthreads();
    }
    const float mu = red[0] * (1.0f / C_M);
    __syncthreads();

    const float dx = x - mu;
    red[t] = dx * dx;
    __syncthreads();
    for (int st = C_M / 2; st > 0; st >>= 1) {
        if (t < st) red[t] += red[t + st];
        __syncthreads();
    }
    const float var = red[0] * (1.0f / C_M);
    __syncthreads();

    const float rs = __frsqrt_rn(var + 1e-5f);
    mnS[t] = dx * rs * ln_w[t] + ln_b[t];
    __syncthreads();

    const int ch  = t & 63;
    const int seg = t >> 6;
    const float* wp = (ch < C_C) ? (w1 + (size_t)ch * C_M)
                                 : (w2 + (size_t)(ch - C_C) * C_M);
    float part = 0.0f;
    #pragma unroll 8
    for (int e = 0; e < 64; ++e) {
        const int idx = seg * 64 + e;
        part += mnS[idx] * wp[idx];
    }
    red[t] = part;
    __syncthreads();

    if (t < 64) {
        float tot = red[t] + red[t + 64] + red[t + 128] + red[t + 192];
        if (t < C_C) {
            tot += b1[t];
            aT[((size_t)i * C_C + t) * N_SEQ + s] = (__bf16)tot;
        } else {
            tot += b2[t - C_C];
            bT[((size_t)i * C_C + (t - C_C)) * N_SEQ + s] = (__bf16)tot;
        }
    }
}

// =====================================================================
// Kernel 1b: convert wo -> bf16 with k-permutation k' = d*32 + c
// (matches the staged-O layout in kernel 2; sum over k is order-invariant)
// =====================================================================
__global__ void opm_cvt_wo_kernel(const float* __restrict__ wo,
                                  __bf16* __restrict__ wo_bf) {
    const int idx = blockIdx.x * blockDim.x + threadIdx.x;
    if (idx < C_Z * KOUT) {
        const int z = idx >> 10;
        const int k = idx & 1023;
        const int c = k >> 5;
        const int d = k & 31;
        wo_bf[((size_t)z << 10) | (d << 5) | c] = (__bf16)wo[idx];
    }
}

// =====================================================================
// Kernel 2: fused  O_ij = A_i^T B_j (K=128)  ->  z = (O.wo^T + bo)/128
// 128 threads (4 waves). block: i = blockIdx.x*4 + wave, j = blockIdx.y*16 + p
// LDS: [bT tile 16j x 32d x 128s padded] [4 x staged 16x1024 bf16 padded]
// =====================================================================
__global__ void opm_wmma_fused_kernel(const __bf16* __restrict__ aT,
                                      const __bf16* __restrict__ bT,
                                      const __bf16* __restrict__ wo_bf,
                                      const float* __restrict__ bo,
                                      float* __restrict__ out) {
    extern __shared__ uint4 smem_u4[];
    char* const ldsBT = (char*)smem_u4;
    char* const ldsA2 = (char*)smem_u4 + BT_TILE_B;

    const int t    = threadIdx.x;
    const int lane = t & 31;
    const int wid  = t >> 5;
    const int hi   = lane >> 4;        // 0: lanes 0-15, 1: lanes 16-31
    const int ln   = lane & 15;

    const int i  = blockIdx.x * 4 + wid;
    const int j0 = blockIdx.y * 16;

    // ---- cooperative stage of bT tile (16 j, contiguous 128KB) into LDS ----
    {
        const uint4* src = (const uint4*)(bT + (size_t)j0 * C_C * N_SEQ); // 8192 uint4
        #pragma unroll 4
        for (int it = 0; it < 64; ++it) {
            const int g   = it * 128 + t;
            const int row = g >> 4;          // 512 rows of 16 uint4
            const int seg = g & 15;
            *(uint4*)(ldsBT + (size_t)row * BT_ROW_B + seg * 16) = src[g];
        }
    }

    // ---- load A_i fragments (32 x 128, rows = c, cols = s), resident ----
    Frag16 afr[2][4];
    #pragma unroll
    for (int ct = 0; ct < 2; ++ct) {
        const int row = i * C_C + ct * 16 + ln;
        const uint4* aRow = (const uint4*)(aT + (size_t)row * N_SEQ); // 16 uint4/row
        #pragma unroll
        for (int kc = 0; kc < 4; ++kc) {
            afr[ct][kc].u[0] = aRow[kc * 4 + hi * 2];
            afr[ct][kc].u[1] = aRow[kc * 4 + hi * 2 + 1];
        }
    }
    __syncthreads();   // bT tile visible

    // ---- phase 1: per pair, O_ij = A_i^T B_j ; stage bf16 (k' = d*32+c) ----
    char* const stagedW = ldsA2 + (size_t)wid * A2_SLICE_B;
    for (int p = 0; p < 16; ++p) {
        v8f o00 = {}, o01 = {}, o10 = {}, o11 = {};
        const char* bRow0 = ldsBT + (size_t)(p * C_C + ln) * BT_ROW_B;
        const char* bRow1 = ldsBT + (size_t)(p * C_C + 16 + ln) * BT_ROW_B;
        #pragma unroll
        for (int kc = 0; kc < 4; ++kc) {
            Frag16 bf0, bf1;
            const int off = kc * 64 + hi * 32;
            bf0.u[0] = *(const uint4*)(bRow0 + off);
            bf0.u[1] = *(const uint4*)(bRow0 + off + 16);
            bf1.u[0] = *(const uint4*)(bRow1 + off);
            bf1.u[1] = *(const uint4*)(bRow1 + off + 16);
            o00 = wmma_bf16(afr[0][kc], bf0, o00);
            o01 = wmma_bf16(afr[0][kc], bf1, o01);
            o10 = wmma_bf16(afr[1][kc], bf0, o10);
            o11 = wmma_bf16(afr[1][kc], bf1, o11);
        }
        // stage O: lane owns column d; r-values contiguous under k' = d*32+c
        char* const sp = stagedW + (size_t)p * A2_ROW_B;
        Pack8 pk0, pk1, pk2, pk3;
        #pragma unroll
        for (int r = 0; r < 8; ++r) {
            pk0.h[r] = (__bf16)o00[r];
            pk1.h[r] = (__bf16)o01[r];
            pk2.h[r] = (__bf16)o10[r];
            pk3.h[r] = (__bf16)o11[r];
        }
        const int base = hi * 16;                       // c offset * 2B
        *(uint4*)(sp + (ln)      * 64 + base)      = pk0.q;  // d=ln,    ct=0
        *(uint4*)(sp + (16 + ln) * 64 + base)      = pk1.q;  // d=16+ln, ct=0
        *(uint4*)(sp + (ln)      * 64 + base + 32) = pk2.q;  // d=ln,    ct=1
        *(uint4*)(sp + (16 + ln) * 64 + base + 32) = pk3.q;  // d=16+ln, ct=1
    }
    __syncthreads();   // all 4 staged tiles visible

    // ---- phase 2: wave owns z-slice (2 z-tiles) over all 4 A-tiles ----
    v8f zacc[8] = {};  // [tile t][zz]
    const int z0 = wid * 2;
    const uint4* wRow0 = (const uint4*)(wo_bf + (size_t)((z0)     * 16 + ln) * KOUT);
    const uint4* wRow1 = (const uint4*)(wo_bf + (size_t)((z0 + 1) * 16 + ln) * KOUT);
    for (int kc = 0; kc < 32; ++kc) {
        Frag16 bz0, bz1;
        bz0.u[0] = wRow0[kc * 4 + hi * 2];
        bz0.u[1] = wRow0[kc * 4 + hi * 2 + 1];
        bz1.u[0] = wRow1[kc * 4 + hi * 2];
        bz1.u[1] = wRow1[kc * 4 + hi * 2 + 1];
        const int offA0 = kc * 64 + hi * 32;
        #pragma unroll
        for (int tt = 0; tt < 4; ++tt) {
            Frag16 a2;
            const char* aRow = ldsA2 + (size_t)tt * A2_SLICE_B + (size_t)ln * A2_ROW_B;
            a2.u[0] = *(const uint4*)(aRow + offA0);
            a2.u[1] = *(const uint4*)(aRow + offA0 + 16);
            zacc[tt * 2]     = wmma_bf16(a2, bz0, zacc[tt * 2]);
            zacc[tt * 2 + 1] = wmma_bf16(a2, bz1, zacc[tt * 2 + 1]);
        }
    }

    // ---- epilogue: out[i(t)][j0+p][z] = (acc + bo[z]) / N_SEQ ----
    #pragma unroll
    for (int zz = 0; zz < 2; ++zz) {
        const int z  = (z0 + zz) * 16 + ln;
        const float bz = bo[z];
        #pragma unroll
        for (int tt = 0; tt < 4; ++tt) {
            const int it = blockIdx.x * 4 + tt;
            #pragma unroll
            for (int r = 0; r < 8; ++r) {
                const int p = hi * 8 + r;
                out[((size_t)it * N_RES + (j0 + p)) * C_Z + z] =
                    (zacc[tt * 2 + zz][r] + bz) * (1.0f / N_SEQ);
            }
        }
    }
}

// =====================================================================
extern "C" void kernel_launch(void* const* d_in, const int* in_sizes, int n_in,
                              void* d_out, int out_size, void* d_ws, size_t ws_size,
                              hipStream_t stream) {
    const float* m    = (const float*)d_in[0];
    const float* ln_w = (const float*)d_in[1];
    const float* ln_b = (const float*)d_in[2];
    const float* w1   = (const float*)d_in[3];
    const float* b1   = (const float*)d_in[4];
    const float* w2   = (const float*)d_in[5];
    const float* b2   = (const float*)d_in[6];
    const float* wo   = (const float*)d_in[7];
    const float* bo   = (const float*)d_in[8];
    float* out = (float*)d_out;

    char* ws = (char*)d_ws;
    __bf16* aT    = (__bf16*)(ws);                         // 2 MB
    __bf16* bT    = (__bf16*)(ws + (2u << 20));            // 2 MB
    __bf16* wo_bf = (__bf16*)(ws + (4u << 20));            // 256 KB

    opm_ln_proj_kernel<<<dim3(N_SEQ, N_RES), 256, 0, stream>>>(
        m, ln_w, ln_b, w1, b1, w2, b2, aT, bT);

    opm_cvt_wo_kernel<<<(C_Z * KOUT + 255) / 256, 256, 0, stream>>>(wo, wo_bf);

    opm_wmma_fused_kernel<<<dim3(N_RES / 4, N_RES / 16), 128, LDS_TOTAL, stream>>>(
        aT, bT, wo_bf, bo, out);
}
